// Adjacency_based_GCN_85856396247988
// MI455X (gfx1250) — matrix operand order repaired and checked by reading, not compiled, not run on hardware
//
#include <hip/hip_runtime.h>
#include <hip/hip_bf16.h>
#include <math.h>
#include <stdint.h>

// ---------------- types ----------------
typedef __attribute__((ext_vector_type(16))) __bf16 v16bf;
typedef __attribute__((ext_vector_type(8)))  __bf16 v8bf;
typedef __attribute__((ext_vector_type(8)))  float  v8f;
typedef __attribute__((ext_vector_type(4))) unsigned int u32x4;
typedef __attribute__((ext_vector_type(8))) int i32x8;
typedef __attribute__((ext_vector_type(4))) int i32x4;

#define EMB 128
#define OUTC 10
#define BN_EPS 1e-5f

// ---------------- TDM: load a 2D bf16 tile [128 rows x 32 cols] into LDS ----------------
// Source tensor: W2t[128][K] bf16 (dim0 = K contiguous). Tile start = column kc of row 0.
// LDS result: contiguous rows of 32 bf16 -> Bs[n*32 + k], exactly the B-fragment layout.
__device__ __forceinline__ void tdm_load_B(unsigned lds_off, const __bf16* gptr, unsigned K) {
    unsigned long long ga = (unsigned long long)(uintptr_t)gptr;
    u32x4 g0;
    g0[0] = 1u;                                            // count=1, user descriptor
    g0[1] = lds_off;                                       // lds_addr
    g0[2] = (unsigned)(ga & 0xFFFFFFFFu);                  // global_addr[31:0]
    g0[3] = (unsigned)((ga >> 32) & 0x01FFFFFFu) | (2u << 30); // global_addr[56:32] | type=2
    i32x8 g1;
    g1[0] = (int)(1u << 16);                               // data_size=2B; wg_mask=0
    g1[1] = (int)((K & 0xFFFFu) << 16);                    // tensor_dim0[15:0] @bits63:48
    g1[2] = (int)(((K >> 16) & 0xFFFFu) | (128u << 16));   // tensor_dim0[31:16] | tensor_dim1[15:0]=128
    g1[3] = (int)(32u << 16);                              // tensor_dim1[31:16]=0 | tile_dim0=32
    g1[4] = (int)128u;                                     // tile_dim1=128, tile_dim2=0
    g1[5] = (int)K;                                        // tensor_dim0_stride[31:0]
    g1[6] = (int)((K & 0xFFFFu) << 16);                    // stride0[47:32]=0 | tensor_dim1_stride[15:0]
    g1[7] = (int)(K >> 16);                                // tensor_dim1_stride[47:16]
    i32x4 z4 = {0, 0, 0, 0};
#if __clang_major__ >= 23
    i32x8 z8 = {0, 0, 0, 0, 0, 0, 0, 0};
    __builtin_amdgcn_tensor_load_to_lds(g0, g1, z4, z4, z8, 0);
#else
    __builtin_amdgcn_tensor_load_to_lds(g0, g1, z4, z4, 0);
#endif
}

// ---------------- utility kernels ----------------
__global__ void fill_f32(float* p, float v, long n) {
    long i = (long)blockIdx.x * blockDim.x + threadIdx.x;
    if (i < n) p[i] = v;
}

// Column sums / sums of squares.  block = 64 threads (2 waves); grid.y covers columns.
__global__ void col_stats(const float* __restrict__ X, int M, int C,
                          float* __restrict__ sums, float* __restrict__ sumsq) {
    int col = blockIdx.y * 64 + threadIdx.x;
    if (col >= C) return;
    int chunk = (M + gridDim.x - 1) / gridDim.x;
    int r0 = blockIdx.x * chunk;
    int r1 = min(M, r0 + chunk);
    float s = 0.f, q = 0.f;
    for (int r = r0; r < r1; ++r) {
        float v = X[(size_t)r * C + col];
        s += v; q += v * v;
    }
    atomicAdd(&sums[col], s);
    atomicAdd(&sumsq[col], q);
}

// s = g * rsqrt(var+eps), t = b - mean*s   (biased var, matches jnp.var)
__global__ void finalize_scale(const float* __restrict__ sums, const float* __restrict__ sumsq,
                               const float* __restrict__ g, const float* __restrict__ b,
                               float invM, float* __restrict__ s, float* __restrict__ t, int C) {
    int i = blockIdx.x * blockDim.x + threadIdx.x;
    if (i >= C) return;
    float m  = sums[i] * invM;
    float v  = sumsq[i] * invM - m * m;
    float si = g[i] * rsqrtf(v + BN_EPS);
    s[i] = si;
    t[i] = b[i] - m * si;
}

// W2t[n*K + k] = (bf16)(s[k] * W[k*128+n])   -- transposed so TDM tiles are contiguous
__global__ void prep_w_bf16_t(const float* __restrict__ W, const float* __restrict__ s,
                              __bf16* __restrict__ W2t, int K) {
    long i = (long)blockIdx.x * blockDim.x + threadIdx.x;
    if (i < (long)K * EMB) {
        int k = (int)(i >> 7);
        int n = (int)(i & 127);
        W2t[(size_t)n * K + k] = (__bf16)(s[k] * W[i]);
    }
}

// out[j] = base[j] + sum_k t[k]*W[k*128+j];  one block (256 thr) per j
__global__ void bias_fold(const float* __restrict__ t, const float* __restrict__ W,
                          const float* __restrict__ base, float* __restrict__ out, int K) {
    __shared__ float red[256];
    int j = blockIdx.x;
    float acc = 0.f;
    for (int k = threadIdx.x; k < K; k += 256) acc += t[k] * W[(size_t)k * EMB + j];
    red[threadIdx.x] = acc;
    __syncthreads();
    for (int sft = 128; sft > 0; sft >>= 1) {
        if (threadIdx.x < sft) red[threadIdx.x] += red[threadIdx.x + sft];
        __syncthreads();
    }
    if (threadIdx.x == 0) out[j] = (base ? base[j] : 0.f) + red[0];
}

// ---------------- WMMA GEMM:  C[M,128] = relu?( [A0|A1][M,K] @ W2t^T + bias ) ----------------
// block = 256 threads = 8 waves; 128-row x 128-col tile per block.
// A: f32 in memory, converted to bf16 while staging to LDS.
// B: pre-transposed bf16 weights W2t[128][K]; each 32xK-chunk tile DMA'd by the TDM.
__global__ __launch_bounds__(256) void gemm_bf16(
    const float* __restrict__ A0, const float* __restrict__ A1, int k0len,
    int K, int M, const __bf16* __restrict__ W2t, const float* __restrict__ bias,
    float* __restrict__ C, int relu)
{
    __shared__ __attribute__((aligned(32))) __bf16 As[128 * 32]; // [row][k]            8KB
    __shared__ __attribute__((aligned(32))) __bf16 Bs[128 * 32]; // [n][k] via TDM      8KB
    const int tid   = threadIdx.x;
    const int wave  = tid >> 5;
    const int lane  = tid & 31;
    const int m0    = blockIdx.x * 128;
    const int srow  = tid >> 1;          // 0..127
    const int shalf = (tid & 1) * 16;    // 0 or 16
    const unsigned bs_lds = (unsigned)(uintptr_t)(void*)&Bs[0];

    v8f acc[8] = {};                     // 8 N-tiles of 16x16 f32

    for (int kc = 0; kc < K; kc += 32) {
        // B tile: one TDM descriptor issued by wave 0 (EXEC-independent, uniform)
        if (wave == 0) {
            tdm_load_B(bs_lds, W2t + kc, (unsigned)K);
        }

        // A source select (concat [A0|A1] along K; chunks never straddle, k0len%32==0)
        const float* Asrc; int lda, kk;
        if (kc < k0len) { Asrc = A0; lda = k0len;     kk = kc;          }
        else            { Asrc = A1; lda = K - k0len; kk = kc - k0len;  }

        // stage A tile (f32 -> bf16)
        {
            int gr = m0 + srow;
            __bf16* adst = &As[srow * 32 + shalf];
            if (gr < M) {
                const float* ap = Asrc + (size_t)gr * lda + kk + shalf;
                if (kc + 32 < K) __builtin_prefetch(ap + 32, 0, 1);
                #pragma unroll
                for (int i = 0; i < 16; ++i) adst[i] = (__bf16)ap[i];
            } else {
                #pragma unroll
                for (int i = 0; i < 16; ++i) adst[i] = (__bf16)0.0f;
            }
        }
        if (wave == 0) __builtin_amdgcn_s_wait_tensorcnt(0);  // TDM tile landed
        __syncthreads();                                       // publish As + Bs

        // A fragment: 16x32 bf16 per wave.  lane<16: K {0..15}; lane>=16: K {16..31}
        const int arow = wave * 16 + (lane & 15);
        const int akb  = (lane < 16) ? 0 : 16;
        v8bf alo = *(const v8bf*)&As[arow * 32 + akb];
        v8bf ahi = *(const v8bf*)&As[arow * 32 + akb + 8];
        v16bf afrag = __builtin_shufflevector(alo, ahi,
                        0,1,2,3,4,5,6,7,8,9,10,11,12,13,14,15);

        // B fragments: lane holds col n=lane&15, 16 consecutive K values
        const int bn  = lane & 15;
        const int bkb = (lane < 16) ? 0 : 16;
        #pragma unroll
        for (int nt = 0; nt < 8; ++nt) {
            v16bf bfrag = *(const v16bf*)&Bs[(nt * 16 + bn) * 32 + bkb];
            acc[nt] = __builtin_amdgcn_wmma_f32_16x16x32_bf16(
                false, afrag, false, bfrag, (short)0, acc[nt], false, false);
        }
        __syncthreads();                                       // all reads done before next tile
    }

    // epilogue: VGPR v holds C[M = v + (lane>=16 ? 8 : 0)][N = lane&15] within the 16x16 tile
    const int rbase = m0 + wave * 16 + ((lane >> 4) << 3);
    const int nlo   = lane & 15;
    #pragma unroll
    for (int nt = 0; nt < 8; ++nt) {
        int col  = nt * 16 + nlo;
        float bv = bias ? bias[col] : 0.f;
        #pragma unroll
        for (int v = 0; v < 8; ++v) {
            int row = rbase + v;
            if (row < M) {
                float val = acc[nt][v] + bv;
                if (relu) val = fmaxf(val, 0.f);
                C[(size_t)row * EMB + col] = val;
            }
        }
    }
}

// ---------------- graph kernels ----------------
__global__ void edge_deg(const int* __restrict__ dst, float* __restrict__ deg, int E) {
    int e = blockIdx.x * blockDim.x + threadIdx.x;
    if (e < E) atomicAdd(&deg[dst[e]], 1.0f);
}

__global__ void deg_rsqrt(const float* __restrict__ deg, float* __restrict__ dinv, int M) {
    int i = blockIdx.x * blockDim.x + threadIdx.x;
    if (i < M) dinv[i] = rsqrtf(deg[i]);   // deg >= 1 always (self-loop)
}

// one wave (32 lanes) per edge; each lane moves 4 of 128 floats via float atomics
__global__ __launch_bounds__(256) void scatter_edges(
    const int* __restrict__ src, const int* __restrict__ dst,
    const float* __restrict__ dinv, const float* __restrict__ XW,
    float* __restrict__ agg, int E)
{
    int wid  = (int)(((long)blockIdx.x * blockDim.x + threadIdx.x) >> 5);
    int lane = threadIdx.x & 31;
    if (wid >= E) return;
    int s = src[wid], d = dst[wid];
    float w = dinv[s] * dinv[d];
    float4 v = ((const float4*)(XW + (size_t)s * EMB))[lane];
    float* ap = agg + (size_t)d * EMB + lane * 4;
    atomicAdd(ap + 0, v.x * w);
    atomicAdd(ap + 1, v.y * w);
    atomicAdd(ap + 2, v.z * w);
    atomicAdd(ap + 3, v.w * w);
}

// one wave per node: add self-loop + conv bias, tanh, 128->10 classifier, log-softmax
__global__ __launch_bounds__(256) void final_head(
    const float* __restrict__ agg, const float* __restrict__ XW,
    const float* __restrict__ dinv, const float* __restrict__ conv_b,
    const float* __restrict__ clsW, const float* __restrict__ clsb,
    float* __restrict__ out, int M)
{
    __shared__ float sW[EMB * OUTC];   // 5KB
    for (int i = threadIdx.x; i < EMB * OUTC; i += 256) sW[i] = clsW[i];
    __syncthreads();

    int node = blockIdx.x * 8 + (threadIdx.x >> 5);
    int lane = threadIdx.x & 31;
    if (node >= M) return;

    float di = dinv[node];
    float d2 = di * di;
    float4 a  = ((const float4*)(agg + (size_t)node * EMB))[lane];
    float4 x  = ((const float4*)(XW  + (size_t)node * EMB))[lane];
    float4 cb = ((const float4*)conv_b)[lane];
    int c0 = lane * 4;
    float y0 = tanhf(a.x + x.x * d2 + cb.x);
    float y1 = tanhf(a.y + x.y * d2 + cb.y);
    float y2 = tanhf(a.z + x.z * d2 + cb.z);
    float y3 = tanhf(a.w + x.w * d2 + cb.w);

    float z[OUTC];
    #pragma unroll
    for (int o = 0; o < OUTC; ++o) {
        float p = y0 * sW[(c0 + 0) * OUTC + o]
                + y1 * sW[(c0 + 1) * OUTC + o]
                + y2 * sW[(c0 + 2) * OUTC + o]
                + y3 * sW[(c0 + 3) * OUTC + o];
        #pragma unroll
        for (int off = 16; off > 0; off >>= 1) p += __shfl_xor(p, off, 32);
        z[o] = p + clsb[o];
    }
    if (lane == 0) {
        float mx = z[0];
        #pragma unroll
        for (int o = 1; o < OUTC; ++o) mx = fmaxf(mx, z[o]);
        float se = 0.f;
        #pragma unroll
        for (int o = 0; o < OUTC; ++o) se += expf(z[o] - mx);
        float lse = mx + logf(se);
        #pragma unroll
        for (int o = 0; o < OUTC; ++o) out[(size_t)node * OUTC + o] = z[o] - lse;
    }
}

// ---------------- launcher ----------------
extern "C" void kernel_launch(void* const* d_in, const int* in_sizes, int n_in,
                              void* d_out, int out_size, void* d_ws, size_t ws_size,
                              hipStream_t stream) {
    (void)n_in; (void)out_size; (void)ws_size;

    const float* Xlow   = (const float*)d_in[1];
    const float* Xhigh  = (const float*)d_in[2];   // cov matrix (high_dim_features unused)
    const int*   ei     = (const int*)  d_in[3];
    const float* bnl_g  = (const float*)d_in[4];
    const float* bnl_b  = (const float*)d_in[5];
    const float* bnh_g  = (const float*)d_in[6];
    const float* bnh_b  = (const float*)d_in[7];
    const float* Wl     = (const float*)d_in[8];
    const float* bl     = (const float*)d_in[9];
    const float* ml_g   = (const float*)d_in[10];
    const float* ml_b   = (const float*)d_in[11];
    const float* Wh     = (const float*)d_in[12];
    const float* bh     = (const float*)d_in[13];
    const float* mh_g   = (const float*)d_in[14];
    const float* mh_b   = (const float*)d_in[15];
    const float* convW  = (const float*)d_in[16];
    const float* convB  = (const float*)d_in[17];
    const float* clsW   = (const float*)d_in[18];
    const float* clsB   = (const float*)d_in[19];
    float* out = (float*)d_out;

    const int N    = in_sizes[1] / 64;         // 50000
    const int HIGH = in_sizes[2] / N;          // 4096
    const int LOW  = in_sizes[1] / N;          // 64
    const int E    = in_sizes[3] / 2;          // 1.6M
    const int* src = ei;
    const int* dst = ei + E;
    const float invN = 1.0f / (float)N;

    // ---- workspace layout ----
    char* base = (char*)d_ws;
    size_t off = 0;
    auto alloc = [&](size_t bytes) -> char* {
        char* p = base + off;
        off = (off + bytes + 255) & ~(size_t)255;
        return p;
    };
    // contiguous zeroed stats region
    float* stats   = (float*)alloc((size_t)(HIGH*2 + LOW*2 + EMB*4) * 4);
    float* sums_h  = stats;
    float* sumsq_h = stats + HIGH;
    float* sums_l  = stats + 2*HIGH;
    float* sumsq_l = stats + 2*HIGH + LOW;
    float* sums_A  = stats + 2*HIGH + 2*LOW;            // relu(high) stats
    float* sumsq_A = sums_A + EMB;
    float* sums_B  = sumsq_A + EMB;                     // relu(low) stats
    float* sumsq_B = sums_B + EMB;
    long   nStats  = HIGH*2 + LOW*2 + EMB*4;

    float*  s_h   = (float*) alloc((size_t)HIGH * 4);
    float*  t_h   = (float*) alloc((size_t)HIGH * 4);
    float*  s_l   = (float*) alloc((size_t)LOW  * 4);
    float*  t_l   = (float*) alloc((size_t)LOW  * 4);
    float*  s3    = (float*) alloc(256 * 4);            // concat(high,low) mlp-bn scales
    float*  t3    = (float*) alloc(256 * 4);
    __bf16* w2h   = (__bf16*)alloc((size_t)HIGH * EMB * 2);   // transposed [128][HIGH]
    __bf16* w2l   = (__bf16*)alloc((size_t)LOW  * EMB * 2);   // transposed [128][LOW]
    __bf16* w3    = (__bf16*)alloc((size_t)256  * EMB * 2);   // transposed [128][256]
    float*  bias2h= (float*) alloc(EMB * 4);
    float*  bias2l= (float*) alloc(EMB * 4);
    float*  bias3 = (float*) alloc(EMB * 4);
    float*  Hh    = (float*) alloc((size_t)N * EMB * 4);
    float*  Hl    = (float*) alloc((size_t)N * EMB * 4);
    float*  XW    = (float*) alloc((size_t)N * EMB * 4);
    float*  deg   = (float*) alloc((size_t)N * 4);
    float*  dinv  = (float*) alloc((size_t)N * 4);
    float*  AGG   = Hh;   // alias: Hh dead after GEMM2

    const int gemmGrid = (N + 127) / 128;

    // 1) zero stats, compute input column stats
    fill_f32<<<(int)((nStats + 255) / 256), 256, 0, stream>>>(stats, 0.f, nStats);
    col_stats<<<dim3(256, HIGH / 64), 64, 0, stream>>>(Xhigh, N, HIGH, sums_h, sumsq_h);
    col_stats<<<dim3(256, 1),         64, 0, stream>>>(Xlow,  N, LOW,  sums_l, sumsq_l);

    // 2) fold input BN into linear weights (bf16, transposed for TDM)
    finalize_scale<<<(HIGH + 255) / 256, 256, 0, stream>>>(sums_h, sumsq_h, bnh_g, bnh_b, invN, s_h, t_h, HIGH);
    finalize_scale<<<1, 64, 0, stream>>>(sums_l, sumsq_l, bnl_g, bnl_b, invN, s_l, t_l, LOW);
    prep_w_bf16_t<<<(int)(((long)HIGH * EMB + 255) / 256), 256, 0, stream>>>(Wh, s_h, w2h, HIGH);
    prep_w_bf16_t<<<(int)(((long)LOW  * EMB + 255) / 256), 256, 0, stream>>>(Wl, s_l, w2l, LOW);
    bias_fold<<<EMB, 256, 0, stream>>>(t_h, Wh, bh, bias2h, HIGH);
    bias_fold<<<EMB, 256, 0, stream>>>(t_l, Wl, bl, bias2l, LOW);

    // 3) WMMA GEMMs with fused ReLU (B tiles via Tensor Data Mover)
    gemm_bf16<<<gemmGrid, 256, 0, stream>>>(Xhigh, nullptr, HIGH, HIGH, N, w2h, bias2h, Hh, 1);
    gemm_bf16<<<gemmGrid, 256, 0, stream>>>(Xlow,  nullptr, LOW,  LOW,  N, w2l, bias2l, Hl, 1);

    // 4) post-ReLU BN stats, fold into conv1_W (combined = [high | low])
    col_stats<<<dim3(256, EMB / 64), 64, 0, stream>>>(Hh, N, EMB, sums_A, sumsq_A);
    col_stats<<<dim3(256, EMB / 64), 64, 0, stream>>>(Hl, N, EMB, sums_B, sumsq_B);
    finalize_scale<<<1, 128, 0, stream>>>(sums_A, sumsq_A, mh_g, mh_b, invN, s3,       t3,       EMB);
    finalize_scale<<<1, 128, 0, stream>>>(sums_B, sumsq_B, ml_g, ml_b, invN, s3 + EMB, t3 + EMB, EMB);
    prep_w_bf16_t<<<(256 * EMB + 255) / 256, 256, 0, stream>>>(convW, s3, w3, 256);
    bias_fold<<<EMB, 256, 0, stream>>>(t3, convW, nullptr, bias3, 256);

    // 5) GCN transform GEMM on concat input
    gemm_bf16<<<gemmGrid, 256, 0, stream>>>(Hh, Hl, EMB, 256, N, w3, bias3, XW, 0);

    // 6) degrees (self-loop => init 1.0) and symmetric norm
    fill_f32<<<(int)(((long)N * EMB + 255) / 256), 256, 0, stream>>>(AGG, 0.f, (long)N * EMB);
    fill_f32<<<(N + 255) / 256, 256, 0, stream>>>(deg, 1.0f, N);
    edge_deg<<<(E + 255) / 256, 256, 0, stream>>>(dst, deg, E);
    deg_rsqrt<<<(N + 255) / 256, 256, 0, stream>>>(deg, dinv, N);

    // 7) scatter messages (wave per edge)
    scatter_edges<<<(E + 7) / 8, 256, 0, stream>>>(src, dst, dinv, XW, AGG, E);

    // 8) self-loop + bias + tanh + classifier + log-softmax (wave per node)
    final_head<<<(N + 7) / 8, 256, 0, stream>>>(AGG, XW, dinv, convB, clsW, clsB, out, N);
}
